// PyGGAT_8632884265054
// MI455X (gfx1250) — matrix operand order repaired and checked by reading, not compiled, not run on hardware
//
#include <hip/hip_runtime.h>
#include <hip/hip_bf16.h>
#include <math.h>

// ---------------------------------------------------------------------------
// GAT (3 layers, 4 heads x 64 feats, 256->256 per layer) for MI455X / gfx1250.
// GEMMs run on v_wmma_f32_16x16x32_f16; graph softmax/scatter passes are the
// memory-bound part (~5-6 GB total -> ~0.25 ms at 23.3 TB/s HBM).
// ---------------------------------------------------------------------------

typedef _Float16 f16;
typedef __attribute__((ext_vector_type(16))) _Float16 v16h;
typedef __attribute__((ext_vector_type(8)))  _Float16 v8h;
typedef __attribute__((ext_vector_type(8)))  float    v8f;

#define COLS 256   // HEADS * HID == HEADS * OUT == 256 for every layer
#define NHEAD 4

// ----------------------------- small helpers -------------------------------

__device__ __forceinline__ float atomic_max_f32(float* addr, float val) {
  // Standard monotone bit-pattern trick: non-negative floats order like ints,
  // negative floats order reversed as uints. emax is initialized to -inf
  // (0xFF800000: very negative int, very large uint) so both paths work.
  if (val >= 0.0f)
    return __int_as_float(atomicMax((int*)addr, __float_as_int(val)));
  return __uint_as_float(atomicMin((unsigned int*)addr, __float_as_uint(val)));
}

__device__ __forceinline__ void edge_sd(const int* __restrict__ ei, int E,
                                        int e, int& s, int& d) {
  if (e < E) { s = ei[e]; d = ei[E + e]; }   // edge_index rows: [0]=src [1]=dst
  else       { s = e - E; d = e - E; }        // appended self-loops
}

__global__ void k_fill(float* __restrict__ p, float v, int n) {
  int i = blockIdx.x * blockDim.x + threadIdx.x;
  if (i < n) p[i] = v;
}

// W: [K=256, Nc=256] f32 row-major  ->  Wt: [Nc=256, K=256] f16 row-major
// (transposed so every WMMA B-fragment lane reads 16 *contiguous* K halves)
__global__ void k_convert_w(const float* __restrict__ W, f16* __restrict__ Wt) {
  int i = blockIdx.x * blockDim.x + threadIdx.x;   // 65536 threads
  int k = i >> 8, n = i & 255;
  Wt[n * 256 + k] = (f16)W[k * 256 + n];
}

// --------------------------- WMMA projection GEMM --------------------------
// H[N,256] = X[N,256] @ W[256,256]; block = 8 waves = 16 rows x 128 cols,
// gridDim.y = 2 covers all 256 output columns. A tile staged in LDS as f16.
__global__ void __launch_bounds__(256) k_gemm(const float* __restrict__ X,
                                              const f16* __restrict__ Wt,
                                              float* __restrict__ H, int N) {
  __shared__ __align__(16) f16 As[16 * 256];       // 16 rows x K=256 (8 KB)
  const int rowBase = blockIdx.x * 16;

  // cooperative load + f32->f16 convert of the A tile (4096 elements)
  const float* src = X + (size_t)rowBase * 256;
  for (int j = 0; j < 16; ++j) {
    int idx = threadIdx.x + j * 256;
    int row = rowBase + (idx >> 8);
    As[idx] = (row < N) ? (f16)src[idx] : (f16)0.0f;
  }
  __syncthreads();

  const int lane = threadIdx.x & 31;
  const int wv   = threadIdx.x >> 5;               // 8 waves per block
  const int lo   = lane & 15, hi = lane >> 4;
  const int colBase = blockIdx.y * 128 + wv * 16;
  const int n    = colBase + lo;

  v8f c = {};
  for (int kk = 0; kk < 8; ++kk) {
    const int k0 = kk * 32;
    // A frag (ISA 16-bit A 16x32): lane holds row M=lo;
    // halves 0..15 = K[k0 + hi*16 ..] (contiguous, matches B)
    v8h a0 = *(const v8h*)(As + lo * 256 + k0 + hi * 16);
    v8h a1 = *(const v8h*)(As + lo * 256 + k0 + hi * 16 + 8);
    v16h a = __builtin_shufflevector(a0, a1, 0, 1, 2, 3, 4, 5, 6, 7,
                                             8, 9, 10, 11, 12, 13, 14, 15);
    // B frag: lane holds B-column n; 16 contiguous K halves at k0 + hi*16
    v16h b = *(const v16h*)(Wt + (size_t)n * 256 + k0 + hi * 16);
    c = __builtin_amdgcn_wmma_f32_16x16x32_f16(false, a, false, b,
                                               (short)0, c, false, false);
  }

  // C/D layout: VGPR v, lane -> C[M = v + 8*hi][Ncol = lo]
  float* dst = H + (size_t)rowBase * 256;
  for (int v = 0; v < 8; ++v) {
    int row = rowBase + 8 * hi + v;
    if (row < N) dst[(size_t)(8 * hi + v) * 256 + n] = c[v];
  }
}

// ------------------- per-node attention logits (wave/node) -----------------
// als[n,h] = sum_f H[n, h*64+f] * a_s[h*64+f]  (a_s flat [4,64] == [256])
__global__ void k_al(const float* __restrict__ H, const float* __restrict__ a_s,
                     const float* __restrict__ a_d, float* __restrict__ als,
                     float* __restrict__ ald, int N) {
  int gid  = blockIdx.x * blockDim.x + threadIdx.x;
  int node = gid >> 5;
  if (node >= N) return;
  int lane = threadIdx.x & 31;
  int base = lane * 8;                         // 8 feats, all inside head lane>>3
  const float* hp = H + (size_t)node * 256 + base;
  float ss = 0.f, sd = 0.f;
  for (int j = 0; j < 8; ++j) {
    float hv = hp[j];
    ss += hv * a_s[base + j];
    sd += hv * a_d[base + j];
  }
  for (int m = 1; m < 8; m <<= 1) {            // reduce within 8-lane head group
    ss += __shfl_xor(ss, m, 32);
    sd += __shfl_xor(sd, m, 32);
  }
  if ((lane & 7) == 0) {
    int h = lane >> 3;
    als[node * NHEAD + h] = ss;
    ald[node * NHEAD + h] = sd;
  }
}

// ------------------------ edge pass 1: segment max -------------------------
__global__ void k_edge_max(const int* __restrict__ ei, int E, int Etot,
                           const float* __restrict__ als,
                           const float* __restrict__ ald,
                           float* __restrict__ emax) {
  int e = blockIdx.x * blockDim.x + threadIdx.x;
  if (e >= Etot) return;
  int s, d; edge_sd(ei, E, e, s, d);
  float4 a = *(const float4*)(als + (size_t)s * 4);
  float4 b = *(const float4*)(ald + (size_t)d * 4);
  float v[4] = {a.x + b.x, a.y + b.y, a.z + b.z, a.w + b.w};
  for (int h = 0; h < 4; ++h) {
    float t = v[h] > 0.f ? v[h] : 0.2f * v[h];      // leaky_relu(., 0.2)
    atomic_max_f32(&emax[(size_t)d * 4 + h], t);
  }
}

// -------------------- edge pass 2: exp + segment sum -----------------------
__global__ void k_edge_exp(const int* __restrict__ ei, int E, int Etot,
                           const float* __restrict__ als,
                           const float* __restrict__ ald,
                           const float* __restrict__ emax,
                           float* __restrict__ ee, float* __restrict__ denom) {
  int e = blockIdx.x * blockDim.x + threadIdx.x;
  if (e >= Etot) return;
  int s, d; edge_sd(ei, E, e, s, d);
  float4 a = *(const float4*)(als + (size_t)s * 4);
  float4 b = *(const float4*)(ald + (size_t)d * 4);
  float4 m = *(const float4*)(emax + (size_t)d * 4);
  float v[4] = {a.x + b.x, a.y + b.y, a.z + b.z, a.w + b.w};
  float mm[4] = {m.x, m.y, m.z, m.w};
  float o[4];
  for (int h = 0; h < 4; ++h) {
    float t = v[h] > 0.f ? v[h] : 0.2f * v[h];
    float mx = isfinite(mm[h]) ? mm[h] : 0.0f;      // mirrors jnp.where(isfinite)
    float x = __expf(t - mx);
    o[h] = x;
    atomicAdd(&denom[(size_t)d * 4 + h], x);
  }
  *(float4*)(ee + (size_t)e * 4) = make_float4(o[0], o[1], o[2], o[3]);
}

// --------------- edge pass 3: weighted scatter-add (wave/edge) -------------
__global__ void k_edge_agg(const int* __restrict__ ei, int E, int Etot,
                           const float* __restrict__ ee,
                           const float* __restrict__ denom,
                           const float* __restrict__ H, float* __restrict__ out) {
  int gid = blockIdx.x * blockDim.x + threadIdx.x;
  int e   = gid >> 5;
  if (e >= Etot) return;
  int lane = threadIdx.x & 31;
  int s, d; edge_sd(ei, E, e, s, d);
  int head = lane >> 3;                       // lane*8 feats sit in one head
  float alpha = ee[(size_t)e * 4 + head] /
                (denom[(size_t)d * 4 + head] + 1e-16f);
  int base = lane * 8;
  const float4* hp = (const float4*)(H + (size_t)s * 256 + base);
  float4 x0 = hp[0], x1 = hp[1];
  float* op = out + (size_t)d * 256 + base;
  atomicAdd(op + 0, alpha * x0.x); atomicAdd(op + 1, alpha * x0.y);
  atomicAdd(op + 2, alpha * x0.z); atomicAdd(op + 3, alpha * x0.w);
  atomicAdd(op + 4, alpha * x1.x); atomicAdd(op + 5, alpha * x1.y);
  atomicAdd(op + 6, alpha * x1.z); atomicAdd(op + 7, alpha * x1.w);
}

__global__ void k_bias(float* __restrict__ out, const float* __restrict__ b,
                       int total) {
  int i = blockIdx.x * blockDim.x + threadIdx.x;
  if (i < total) out[i] += b[i & 255];
}

// ------------------------------- host side ---------------------------------

static void run_layer(const float* Xin, const float* W, const float* a_s,
                      const float* a_d, const float* bias, const int* ei,
                      float* P /*proj*/, f16* Wt, float* als, float* ald,
                      float* emax, float* denom, float* ee,
                      float* Oagg /*layer output*/, int N, int E,
                      hipStream_t st) {
  const int Etot = E + N;
  const int TB = 256;
  k_convert_w<<<(256 * 256) / TB, TB, 0, st>>>(W, Wt);
  // GEMM must read Xin BEFORE Oagg is zeroed (Xin may alias Oagg)
  k_gemm<<<dim3((N + 15) / 16, 2), TB, 0, st>>>(Xin, Wt, P, N);
  k_al<<<(N * 32 + TB - 1) / TB, TB, 0, st>>>(P, a_s, a_d, als, ald, N);
  k_fill<<<(N * 4 + TB - 1) / TB, TB, 0, st>>>(emax, -INFINITY, N * 4);
  k_fill<<<(N * 4 + TB - 1) / TB, TB, 0, st>>>(denom, 0.0f, N * 4);
  k_fill<<<(N * 256 + TB - 1) / TB, TB, 0, st>>>(Oagg, 0.0f, N * 256);
  k_edge_max<<<(Etot + TB - 1) / TB, TB, 0, st>>>(ei, E, Etot, als, ald, emax);
  k_edge_exp<<<(Etot + TB - 1) / TB, TB, 0, st>>>(ei, E, Etot, als, ald, emax,
                                                  ee, denom);
  k_edge_agg<<<(Etot * 32 + TB - 1) / TB, TB, 0, st>>>(ei, E, Etot, ee, denom,
                                                       P, Oagg);
  k_bias<<<(N * 256 + TB - 1) / TB, TB, 0, st>>>(Oagg, bias, N * 256);
}

extern "C" void kernel_launch(void* const* d_in, const int* in_sizes, int n_in,
                              void* d_out, int out_size, void* d_ws,
                              size_t ws_size, hipStream_t stream) {
  const float* x   = (const float*)d_in[0];
  const int*   ei  = (const int*)d_in[1];
  const float* W0  = (const float*)d_in[2];
  const float* as0 = (const float*)d_in[3];
  const float* ad0 = (const float*)d_in[4];
  const float* b0  = (const float*)d_in[5];
  const float* W1  = (const float*)d_in[6];
  const float* as1 = (const float*)d_in[7];
  const float* ad1 = (const float*)d_in[8];
  const float* b1  = (const float*)d_in[9];
  const float* W2  = (const float*)d_in[10];
  const float* as2 = (const float*)d_in[11];
  const float* ad2 = (const float*)d_in[12];
  const float* b2  = (const float*)d_in[13];

  const int N = in_sizes[0] / 256;     // 50000
  const int E = in_sizes[1] / 2;       // 800000
  const int Etot = E + N;

  // workspace carve (256B aligned): ~119 MB total
  char* w = (char*)d_ws;
  auto take = [&](size_t bytes) -> void* {
    void* p = (void*)w;
    w += (bytes + 255) & ~(size_t)255;
    return p;
  };
  float* P     = (float*)take((size_t)N * 256 * 4);   // projection h
  float* Q     = (float*)take((size_t)N * 256 * 4);   // layer output ping-pong
  float* als   = (float*)take((size_t)N * 4 * 4);
  float* ald   = (float*)take((size_t)N * 4 * 4);
  float* emax  = (float*)take((size_t)N * 4 * 4);
  float* denom = (float*)take((size_t)N * 4 * 4);
  float* ee    = (float*)take((size_t)Etot * 4 * 4);
  f16*   Wt    = (f16*)take((size_t)256 * 256 * 2);
  (void)n_in; (void)out_size; (void)ws_size;

  float* out = (float*)d_out;
  run_layer(x, W0, as0, ad0, b0, ei, P, Wt, als, ald, emax, denom, ee, Q,
            N, E, stream);                              // layer 0: x -> Q
  run_layer(Q, W1, as1, ad1, b1, ei, P, Wt, als, ald, emax, denom, ee, Q,
            N, E, stream);                              // layer 1: Q -> Q
  run_layer(Q, W2, as2, ad2, b2, ei, P, Wt, als, ald, emax, denom, ee, out,
            N, E, stream);                              // layer 2: Q -> d_out
}
